// PoisSpike_GLM_54082228191292
// MI455X (gfx1250) — matrix operand order repaired and checked by reading, not compiled, not run on hardware
//
#include <hip/hip_runtime.h>

#define T_DATA 50000
#define E_NO   1000
#define I_NO   200
#define SUB    20
#define TNO    200
#define NB     13

typedef __attribute__((ext_vector_type(16))) _Float16 v16h;
typedef __attribute__((ext_vector_type(8)))  float    v8f;
typedef __attribute__((ext_vector_type(4)))  float    f4;

// ---------------------------------------------------------------------------
// Kernel 0: raised-cosine basis + 4 filter banks (e/i/hist/prop), 1 block.
// kern_ws layout: [f][s][t], f=0:e 1:i 2:hist 3:prop  (idx = f*4000 + s*200 + t)
// ---------------------------------------------------------------------------
__global__ void basis_kernel(const float* __restrict__ W_syn,
                             const float* __restrict__ W_hist,
                             const float* __restrict__ W_prop,
                             float* kern_ws, float* filt_out) {
  __shared__ float basis[NB * TNO];
  const int tid = threadIdx.x;
  for (int idx = tid; idx < NB * TNO; idx += blockDim.x) {
    int i = idx / TNO, x = idx % TNO;
    float raw = 5.f * logf((float)x + 1.f + 1e-8f);
    float phi = 1.57079632679f * (float)i;
    float d   = raw - phi;
    float v   = 0.5f * cosf(d) + 0.5f;
    basis[idx] = (d >= -3.14159265358979f && d <= 3.14159265358979f) ? v : 0.f;
  }
  __syncthreads();
  for (int idx = tid; idx < 4 * SUB * TNO; idx += blockDim.x) {
    int f   = idx / (SUB * TNO);
    int rem = idx % (SUB * TNO);
    int s   = rem / TNO, t = rem % TNO;
    float acc = 0.f;
    for (int i = 0; i < NB; ++i) {
      float w;
      if      (f == 0) w = W_syn[(s * NB + i) * 2 + 0];
      else if (f == 1) w = W_syn[(s * NB + i) * 2 + 1];
      else if (f == 2) w = W_hist[s * NB + i];
      else             w = W_prop[s * NB + i];
      acc = fmaf(w, basis[i * TNO + t], acc);
    }
    kern_ws[idx]  = acc;
    filt_out[idx] = acc;
  }
}

// ---------------------------------------------------------------------------
// Kernel 1: pack B = C^T (padded [K x 32]) into WMMA 16x16x32 B register
// layout: Bp[((kc*2+nt)*32 + lane)*16 + j]; K = kc*32 + (j>=8 ? 16 : 0)
// + (lane>=16 ? 8 : 0) + (j&7), N = nt*16 + (lane&15).
// ---------------------------------------------------------------------------
__global__ void pack_b(const float* __restrict__ C /*[SUB][K]*/,
                       _Float16* Bp, int K, int nkc) {
  int total = nkc * 2 * 32 * 16;
  for (int idx = blockIdx.x * blockDim.x + threadIdx.x; idx < total;
       idx += gridDim.x * blockDim.x) {
    int j    = idx & 15;
    int lane = (idx >> 4) & 31;
    int nt   = (idx >> 9) & 1;
    int kc   = idx >> 10;
    int k = kc * 32 + ((j & 8) ? 16 : 0) + ((lane & 16) ? 8 : 0) + (j & 7);
    int n = nt * 16 + (lane & 15);
    float v = 0.f;
    if (k < K && n < SUB) v = C[n * K + k];
    Bp[idx] = (_Float16)v;
  }
}

// Full (unguarded) 32-K A-fragment load; rows are 16B-aligned, S streamed once
// -> non-temporal so the 192MB L2 keeps the reused buffers instead.
__device__ __forceinline__ v16h load_a32(const float* __restrict__ p) {
  const f4* q0 = reinterpret_cast<const f4*>(p);
  const f4* q1 = reinterpret_cast<const f4*>(p + 16);
  f4 x0 = __builtin_nontemporal_load(q0);
  f4 x1 = __builtin_nontemporal_load(q0 + 1);
  f4 x2 = __builtin_nontemporal_load(q1);
  f4 x3 = __builtin_nontemporal_load(q1 + 1);
  v16h a;
  a[0]  = (_Float16)x0[0]; a[1]  = (_Float16)x0[1];
  a[2]  = (_Float16)x0[2]; a[3]  = (_Float16)x0[3];
  a[4]  = (_Float16)x1[0]; a[5]  = (_Float16)x1[1];
  a[6]  = (_Float16)x1[2]; a[7]  = (_Float16)x1[3];
  a[8]  = (_Float16)x2[0]; a[9]  = (_Float16)x2[1];
  a[10] = (_Float16)x2[2]; a[11] = (_Float16)x2[3];
  a[12] = (_Float16)x3[0]; a[13] = (_Float16)x3[1];
  a[14] = (_Float16)x3[2]; a[15] = (_Float16)x3[3];
  return a;
}

// ---------------------------------------------------------------------------
// Kernel 2: syn = S @ C^T via v_wmma_f32_16x16x32_f16 (exact: 0/1 data).
// One wave per 16-row M tile; A reused for both N tiles; software-pipelined
// full-chunk loop + peeled ragged tail.
// ---------------------------------------------------------------------------
__global__ void gemm_spike(const float* __restrict__ S,
                           const _Float16* __restrict__ Bp,
                           float* __restrict__ out,
                           int K, int nkc, int ntiles) {
  const int lane  = threadIdx.x & 31;
  const int tileM = blockIdx.x * (blockDim.x >> 5) + (threadIdx.x >> 5);
  if (tileM >= ntiles) return;

  const int row = tileM * 16 + (lane & 15);
  const float* srow = S + (size_t)row * K;
  const int kofflo = (lane & 16) ? 8 : 0;
  const v16h* bp = reinterpret_cast<const v16h*>(Bp);
  const int nfull = K >> 5;                       // chunks with all 32 k valid

  v8f acc0 = {}, acc1 = {};
  if (nfull > 0) {
    v16h a  = load_a32(srow + kofflo);
    v16h b0 = bp[lane];
    v16h b1 = bp[32 + lane];
    for (int kc = 1; kc < nfull; ++kc) {
      v16h an  = load_a32(srow + kc * 32 + kofflo);
      v16h b0n = bp[(kc * 2 + 0) * 32 + lane];
      v16h b1n = bp[(kc * 2 + 1) * 32 + lane];
      acc0 = __builtin_amdgcn_wmma_f32_16x16x32_f16(false, a, false, b0,
                                                    (short)0, acc0, false, false);
      acc1 = __builtin_amdgcn_wmma_f32_16x16x32_f16(false, a, false, b1,
                                                    (short)0, acc1, false, false);
      a = an; b0 = b0n; b1 = b1n;
    }
    acc0 = __builtin_amdgcn_wmma_f32_16x16x32_f16(false, a, false, b0,
                                                  (short)0, acc0, false, false);
    acc1 = __builtin_amdgcn_wmma_f32_16x16x32_f16(false, a, false, b1,
                                                  (short)0, acc1, false, false);
  }
  if (nfull < nkc) {                              // ragged last K-chunk
    const int kb = nfull * 32 + kofflo;
    v16h a;
#pragma unroll
    for (int j = 0; j < 8; ++j) {
      int k0 = kb + j, k1 = kb + 16 + j;
      a[j]     = (k0 < K) ? (_Float16)srow[k0] : (_Float16)0.f;
      a[8 + j] = (k1 < K) ? (_Float16)srow[k1] : (_Float16)0.f;
    }
    v16h b0 = bp[(nfull * 2 + 0) * 32 + lane];
    v16h b1 = bp[(nfull * 2 + 1) * 32 + lane];
    acc0 = __builtin_amdgcn_wmma_f32_16x16x32_f16(false, a, false, b0,
                                                  (short)0, acc0, false, false);
    acc1 = __builtin_amdgcn_wmma_f32_16x16x32_f16(false, a, false, b1,
                                                  (short)0, acc1, false, false);
  }
  // D layout: VGPR r, lane L -> M = r + (L>=16 ? 8 : 0), N = L&15
#pragma unroll
  for (int r = 0; r < 8; ++r) {
    int m = r + ((lane & 16) ? 8 : 0);
    float* orow = out + (size_t)(tileM * 16 + m) * SUB;
    int n0 = lane & 15;
    orow[n0] = acc0[r];
    if (n0 + 16 < SUB) orow[n0 + 16] = acc1[r];
  }
}

// ---------------------------------------------------------------------------
// Kernel 3: depthwise causal filter: syn[t,s] = sum_tau e_k[s,tau]*syn_e[t-tau,s]
//                                             + i_k[s,tau]*syn_i[t-tau,s]
// ---------------------------------------------------------------------------
__global__ void conv_kernel(const float* __restrict__ syn_e,
                            const float* __restrict__ syn_i,
                            const float* __restrict__ kern_ei /*8000 f32*/,
                            float* __restrict__ syn_out) {
  __shared__ float sk[2 * SUB * TNO];
  for (int i = threadIdx.x; i < 2 * SUB * TNO; i += blockDim.x)
    sk[i] = kern_ei[i];
  __syncthreads();
  int idx = blockIdx.x * blockDim.x + threadIdx.x;
  if (idx >= T_DATA * SUB) return;
  int t = idx / SUB, s = idx % SUB;
  const float* ek = sk + s * TNO;
  const float* ik = sk + SUB * TNO + s * TNO;
  float acc = 0.f;
  int taumax = (t < TNO - 1) ? t : (TNO - 1);
  for (int tau = 0; tau <= taumax; ++tau)
    acc = fmaf(ek[tau], syn_e[(size_t)(t - tau) * SUB + s], acc);
  for (int tau = 0; tau <= taumax; ++tau)
    acc = fmaf(ik[tau], syn_i[(size_t)(t - tau) * SUB + s], acc);
  syn_out[idx] = acc;
}

// ---------------------------------------------------------------------------
// Kernel 4: sequential GLM scan, event-driven. Single wave32.
// F[256][SUB] LDS ring buffer of future hist+prop drive. syn aliases L_out
// (read slot t, then overwrite slot t with L). LDS ops are in-order per wave.
// ---------------------------------------------------------------------------
__global__ void scan_kernel(const float* syn,
                            const float* __restrict__ hk_g,
                            const float* __restrict__ pk_g,
                            const float* __restrict__ Theta,
                            const float* __restrict__ C_den,
                            float* Z_out, float* L_out) {
  __shared__ float F[256 * SUB];
  __shared__ float hk[SUB * TNO];
  __shared__ float pk[SUB * TNO];
  __shared__ float th[SUB];
  __shared__ int   par[SUB];
  const int tid = threadIdx.x;
  for (int i = tid; i < 256 * SUB; i += blockDim.x) F[i] = 0.f;
  for (int i = tid; i < SUB * TNO; i += blockDim.x) { hk[i] = hk_g[i]; pk[i] = pk_g[i]; }
  if (tid < SUB) {
    th[tid] = Theta[tid];
    int p = -1;
    for (int r = 0; r < SUB; ++r)
      if (C_den[r * SUB + tid] != 0.f) p = r;   // single parent per child
    par[tid] = p;
  }
  __syncthreads();
  if (tid >= 32) return;                 // one wave carries the recurrence

  const int  s      = tid;
  const bool active = (s < SUB);
  const float ths   = active ? th[s] : 0.f;
  const int   ps    = active ? par[s] : -1;
  float syn_cur = active ? syn[s] : 0.f;

  for (int t = 0; t < T_DATA; ++t) {
    // deep prefetch (global_prefetch_b8) + 1-step register prefetch
    if (active) {
      __builtin_prefetch(syn + (size_t)(t + 8) * SUB + s, 0, 1);
    }
    float syn_nxt = 0.f;
    if (active && t + 1 < T_DATA) syn_nxt = syn[(size_t)(t + 1) * SUB + s];

    float z = 0.f, L = 0.f;
    const int slot = (t & 255) * SUB;
    if (active) {
      float f = F[slot + s];
      F[slot + s] = 0.f;                 // recycle for step t+256
      L = __expf(syn_cur + ths + f);
      z = rintf(L);                      // round-half-even, matches jnp.round
      L_out[(size_t)t * SUB + s] = L;    // overwrites syn[t,s] after use
      Z_out[(size_t)t * SUB + s] = z;
    }
    if (__ballot(z != 0.f)) {            // rare: scatter spike into the future
      if (active && z != 0.f) {
        const float* hks = hk + s * TNO;
        const float* pks = pk + s * TNO;
        for (int d = 1; d <= TNO; ++d) {
          int sl = ((t + d) & 255) * SUB;
          atomicAdd(&F[sl + s], hks[d - 1] * z);
          if (ps >= 0) atomicAdd(&F[sl + ps], pks[d - 1] * z);
        }
      }
    }
    syn_cur = syn_nxt;
  }
}

// ---------------------------------------------------------------------------
extern "C" void kernel_launch(void* const* d_in, const int* in_sizes, int n_in,
                              void* d_out, int out_size, void* d_ws, size_t ws_size,
                              hipStream_t stream) {
  const float* S_e     = (const float*)d_in[0];
  const float* S_i     = (const float*)d_in[1];
  const float* C_den   = (const float*)d_in[2];
  const float* C_syn_e = (const float*)d_in[3];
  const float* C_syn_i = (const float*)d_in[4];
  const float* W_syn   = (const float*)d_in[5];
  const float* W_hist  = (const float*)d_in[6];
  const float* W_prop  = (const float*)d_in[7];
  const float* Theta   = (const float*)d_in[8];

  float* Z_out    = (float*)d_out;                 // [50000,20]
  float* L_out    = Z_out + T_DATA * SUB;          // [50000,20] (also syn staging)
  float* filt_out = Z_out + 2 * T_DATA * SUB;      // [80,200]

  float*    ws_f  = (float*)d_ws;
  float*    kern  = ws_f;                          // 16000 f32: e,i,hist,prop
  _Float16* Bp_e  = (_Float16*)(ws_f + 16000);     // 32768 halves (32B aligned)
  _Float16* Bp_i  = Bp_e + 32768;                  // 7168 halves
  float*    syn_e = ws_f + 16000 + 16384 + 3584;   // [50000,20]
  float*    syn_i = syn_e + T_DATA * SUB;          // [50000,20]

  const int nkc_e = (E_NO + 31) / 32;              // 32
  const int nkc_i = (I_NO + 31) / 32;              // 7
  const int ntiles = T_DATA / 16;                  // 3125 (exact)

  basis_kernel<<<1, 256, 0, stream>>>(W_syn, W_hist, W_prop, kern, filt_out);
  pack_b<<<64, 256, 0, stream>>>(C_syn_e, Bp_e, E_NO, nkc_e);
  pack_b<<<16, 256, 0, stream>>>(C_syn_i, Bp_i, I_NO, nkc_i);
  gemm_spike<<<(ntiles + 7) / 8, 256, 0, stream>>>(S_e, Bp_e, syn_e, E_NO, nkc_e, ntiles);
  gemm_spike<<<(ntiles + 7) / 8, 256, 0, stream>>>(S_i, Bp_i, syn_i, I_NO, nkc_i, ntiles);
  conv_kernel<<<(T_DATA * SUB + 255) / 256, 256, 0, stream>>>(syn_e, syn_i, kern, L_out);
  scan_kernel<<<1, 256, 0, stream>>>(L_out, kern + 2 * SUB * TNO, kern + 3 * SUB * TNO,
                                     Theta, C_den, Z_out, L_out);
}
